// Decoder_45921790329638
// MI455X (gfx1250) — compile-verified, matrix operand order checked by reading, not run on hardware
//
#include <hip/hip_runtime.h>
#include <hip/hip_bf16.h>
#include <stdint.h>

// ---------------- sizes ----------------
#define B       256
#define EMB     300
#define HID     700
#define ZDIM    500
#define AEMB    200
#define VOCAB   50257
#define MAXLEN  30
#define SOS_IDX 1
#define EOS_IDX 2

#define KX      320            // EMB padded to mult of 32
#define KH      704            // HID padded to mult of 32
#define NG      2100           // 3*HID
#define NGP     2112           // padded to mult of 16 (132 tiles)
#define NGT     132
#define NVT     3142           // ceil(VOCAB/16)
#define NVP     (NVT*16)       // 50272
#define SEQ     (MAXLEN+1)     // 31

#if __has_builtin(__builtin_amdgcn_global_load_async_to_lds_b128) && \
    __has_builtin(__builtin_amdgcn_s_wait_asynccnt)
#define HAVE_ASYNC_LDS 1
#endif

typedef __attribute__((ext_vector_type(16))) _Float16 v16h;
typedef __attribute__((ext_vector_type(8)))  float    v8f;
typedef int v4i_ __attribute__((vector_size(4 * sizeof(int))));
typedef __attribute__((address_space(1))) v4i_* gv4p;   // global-AS 16B vector ptr
typedef __attribute__((address_space(3))) v4i_* lv4p;   // LDS-AS 16B vector ptr

// ---------------- threefry2x32-20 (JAX) ----------------
__host__ __device__ inline void tf2x32(unsigned k0, unsigned k1,
                                       unsigned x0, unsigned x1,
                                       unsigned& y0, unsigned& y1) {
  unsigned ks[3] = {k0, k1, k0 ^ k1 ^ 0x1BD11BDAu};
  const unsigned rot[8] = {13u,15u,26u,6u,17u,29u,16u,24u};
  x0 += ks[0]; x1 += ks[1];
  #pragma unroll
  for (int g = 0; g < 5; ++g) {
    #pragma unroll
    for (int i = 0; i < 4; ++i) {
      unsigned r = rot[(g & 1) * 4 + i];
      x0 += x1;
      x1 = (x1 << r) | (x1 >> (32u - r));
      x1 ^= x0;
    }
    x0 += ks[(g + 1) % 3];
    x1 += ks[(g + 2) % 3] + (unsigned)(g + 1);
  }
  y0 = x0; y1 = x1;
}

// JAX uniform(minval=tiny,maxval=1) -> gumbel = -log(-log(u)).
// Flattened counts iota(B*VOCAB), pair (i, i+m), out[i]= i<m ? y0 : y1.
__device__ inline float gumbel_at(unsigned k0, unsigned k1, unsigned i) {
  const unsigned m = (unsigned)((size_t)B * VOCAB / 2); // 6432896
  unsigned x0, x1, y0, y1;
  if (i < m) { x0 = i; x1 = i + m; } else { x0 = i - m; x1 = i; }
  tf2x32(k0, k1, x0, x1, y0, y1);
  unsigned bits = (i < m) ? y0 : y1;
  float u01 = __uint_as_float((bits >> 9) | 0x3f800000u) - 1.0f;
  const float tiny = 1.175494350822288e-38f;
  float u = fmaxf(tiny, u01 * (1.0f - tiny) + tiny);
  return -logf(-logf(u));
}

__device__ inline unsigned ford(float v) {           // order-preserving f32->u32
  unsigned u = __float_as_uint(v);
  return (u & 0x80000000u) ? ~u : (u | 0x80000000u);
}

__device__ inline unsigned long long shflx64(unsigned long long v, int mask) {
  unsigned lo = (unsigned)v, hi = (unsigned)(v >> 32);
  lo = (unsigned)__shfl_xor((int)lo, mask, 32);
  hi = (unsigned)__shfl_xor((int)hi, mask, 32);
  return ((unsigned long long)hi << 32) | lo;
}

// Load one 16x32 f16 fragment slice for this lane from a row-major [*,Kpad]
// matrix. elems 0..7 -> K = kk+8*hi .. +7 ; elems 8..15 -> K = kk+16+8*hi .. +7
__device__ inline v16h load_frag(const _Float16* p) {
  v16h r;
  __builtin_memcpy(&r, p, 16);
  __builtin_memcpy((char*)&r + 16, p + 16, 16);
  return r;
}

__device__ inline v8f wmma_f16(v16h a, v16h b, v8f c) {
  return __builtin_amdgcn_wmma_f32_16x16x32_f16(false, a, false, b,
                                                (short)0, c, false, false);
}

// ---------------- one-time kernels ----------------
__global__ void conv_w_k(const float* __restrict__ src, _Float16* __restrict__ dst,
                         int N, int K, int Kpad, size_t tot) {
  size_t idx = (size_t)blockIdx.x * blockDim.x + threadIdx.x;
  if (idx >= tot) return;
  int n = (int)(idx / Kpad), k = (int)(idx % Kpad);
  dst[idx] = (n < N && k < K) ? (_Float16)src[(size_t)n * K + k] : (_Float16)0.0f;
}

__global__ void init_h_k(const float* __restrict__ z, const int* __restrict__ l,
                         const float* __restrict__ attr_w,
                         float* h32, _Float16* h16, int* tok) {
  int idx = blockIdx.x * blockDim.x + threadIdx.x;   // B*KH
  if (idx >= B * KH) return;
  int b = idx / KH, j = idx % KH;
  float v = 0.0f;
  if (j < ZDIM)      v = z[b * ZDIM + j];
  else if (j < HID)  v = attr_w[l[b] * AEMB + (j - ZDIM)];
  h16[idx] = (_Float16)v;
  if (j < HID) h32[b * HID + j] = v;
  if (idx < B) tok[idx] = SOS_IDX;
}

// ---------------- per-step kernels ----------------
__global__ void gather_x_k(const float* __restrict__ emb, const int* __restrict__ tok,
                           _Float16* __restrict__ x16, int fixed_tok) {
  int idx = blockIdx.x * blockDim.x + threadIdx.x;   // B*KX
  if (idx >= B * KX) return;
  int b = idx / KX, k = idx % KX;
  int t = (fixed_tok >= 0) ? fixed_tok : tok[b];
  x16[idx] = (k < EMB) ? (_Float16)emb[(size_t)t * EMB + k] : (_Float16)0.0f;
}

// C[B][Ntiles*16] f32 = A[B][Kpad] f16 x W[Ntiles*16][Kpad]^T, one 16x16 tile/wave
__global__ void gemm_k(const _Float16* __restrict__ A, const _Float16* __restrict__ W,
                       float* __restrict__ C, int Kpad, int Ntiles) {
  int wave = blockIdx.x * (blockDim.x >> 5) + (threadIdx.x >> 5);
  if (wave >= (B / 16) * Ntiles) return;             // wave-uniform
  int lane = threadIdx.x & 31;
  int hi = lane >> 4, l16 = lane & 15;
  int mt = wave / Ntiles, nt = wave % Ntiles;
  const _Float16* arow = A + (size_t)(mt * 16 + l16) * Kpad + 8 * hi;
  const _Float16* brow = W + (size_t)(nt * 16 + l16) * Kpad + 8 * hi;
  v8f acc = {};
  for (int kk = 0; kk < Kpad; kk += 32)
    acc = wmma_f16(load_frag(arow + kk), load_frag(brow + kk), acc);
  int Npad = Ntiles * 16;
  int n = nt * 16 + l16;
  #pragma unroll
  for (int j = 0; j < 8; ++j)
    C[(size_t)(mt * 16 + j + 8 * hi) * Npad + n] = acc[j];
}

__global__ void gates_k(const float* __restrict__ gi, const float* __restrict__ gh,
                        const float* __restrict__ b_ih, const float* __restrict__ b_hh,
                        const float* __restrict__ h_prev,
                        float* __restrict__ h_new, _Float16* __restrict__ h16,
                        float* __restrict__ hy, int t,
                        unsigned long long* __restrict__ packed) {
  int idx = blockIdx.x * blockDim.x + threadIdx.x;   // B*HID
  if (idx < B) packed[idx] = 0ull;                   // reset argmax atoms
  if (idx >= B * HID) return;
  int b = idx / HID, j = idx % HID;
  size_t r = (size_t)b * NGP;
  float ir = gi[r + j]           + b_ih[j];
  float iz = gi[r + HID + j]     + b_ih[HID + j];
  float in_= gi[r + 2*HID + j]   + b_ih[2*HID + j];
  float hr = gh[r + j]           + b_hh[j];
  float hz = gh[r + HID + j]     + b_hh[HID + j];
  float hn = gh[r + 2*HID + j]   + b_hh[2*HID + j];
  float rg = 1.0f / (1.0f + expf(-(ir + hr)));
  float zg = 1.0f / (1.0f + expf(-(iz + hz)));
  float ng = tanhf(in_ + rg * hn);
  float h  = (1.0f - zg) * ng + zg * h_prev[idx];
  h_new[idx] = h;
  h16[(size_t)b * KH + j] = (_Float16)h;
  hy[((size_t)b * SEQ + t) * HID + j] = h;
}

// logits = h@out_w^T + out_b; fused Gumbel-argmax (jax.random.categorical).
// grid = (ceil(NVT/8), 4 row-groups), block = 128 (4 waves).
// Each block stages a 64-row A-strip in LDS (dynamic, 90112 B; async global->LDS
// copy when the toolchain exposes it); each wave runs 2 col tiles and reuses
// every B fragment across 4 row-tile accumulators (4x less L2 traffic on the
// 70 MB out_w stream).
__global__ void logits_sample_k(const _Float16* __restrict__ H,
                                const _Float16* __restrict__ Wout,
                                const float* __restrict__ out_b,
                                unsigned long long* __restrict__ packed,
                                unsigned key0, unsigned key1) {
  extern __shared__ _Float16 As[];                   // 64 * KH halves
  int rg = blockIdx.y;                               // 64-row group
#ifdef HAVE_ASYNC_LDS
  {
    gv4p gsrc = (gv4p)(H + (size_t)rg * 64 * KH);    // 16B granules
    lv4p ldst = (lv4p)As;
    for (int i = threadIdx.x; i < (64 * KH * 2) / 16; i += blockDim.x)
      __builtin_amdgcn_global_load_async_to_lds_b128(gsrc + i, ldst + i, 0, 0);
    __builtin_amdgcn_s_wait_asynccnt(0);
  }
#else
  {
    const float4* src = (const float4*)(H + (size_t)rg * 64 * KH);
    float4* dst = (float4*)As;
    for (int i = threadIdx.x; i < 64 * KH / 8; i += blockDim.x) dst[i] = src[i];
  }
#endif
  __syncthreads();

  int wave = threadIdx.x >> 5, lane = threadIdx.x & 31;
  int hi = lane >> 4, l16 = lane & 15;
  const _Float16* abase = As + (size_t)l16 * KH + 8 * hi;

  unsigned long long best[4][8];
  #pragma unroll
  for (int r = 0; r < 4; ++r)
    #pragma unroll
    for (int j = 0; j < 8; ++j) best[r][j] = 0ull;

  for (int c = wave; c < 8; c += 4) {
    int ct = blockIdx.x * 8 + c;
    if (ct >= NVT) break;                            // wave-uniform
    const _Float16* brow = Wout + (size_t)(ct * 16 + l16) * KH + 8 * hi;
    if (ct + 4 < NVT)                                // prefetch next tile's B row
      __builtin_prefetch(Wout + (size_t)((ct + 4) * 16 + l16) * KH, 0, 1);
    v8f acc[4] = {{}, {}, {}, {}};
    for (int kk = 0; kk < KH; kk += 32) {
      v16h bf = load_frag(brow + kk);
      #pragma unroll
      for (int r = 0; r < 4; ++r) {
        v16h af = load_frag(abase + (size_t)r * 16 * KH + kk);
        acc[r] = wmma_f16(af, bf, acc[r]);
      }
    }
    int n = ct * 16 + l16;
    bool ok = (n < VOCAB);
    float bias = ok ? out_b[n] : 0.0f;
    #pragma unroll
    for (int r = 0; r < 4; ++r) {
      #pragma unroll
      for (int j = 0; j < 8; ++j) {
        int row = rg * 64 + r * 16 + j + 8 * hi;     // batch index
        float v = -3.4e38f;
        if (ok) {
          unsigned i = (unsigned)row * (unsigned)VOCAB + (unsigned)n;
          v = acc[r][j] + bias + gumbel_at(key0, key1, i);
        }
        unsigned long long pk = ((unsigned long long)ford(v) << 32)
                              | (unsigned long long)(0xFFFFFFFFu - (unsigned)n);
        if (pk > best[r][j]) best[r][j] = pk;
      }
    }
  }
  // reduce over the 16 lanes of each half (same set of rows per half)
  #pragma unroll
  for (int mask = 1; mask <= 8; mask <<= 1)
    #pragma unroll
    for (int r = 0; r < 4; ++r)
      #pragma unroll
      for (int j = 0; j < 8; ++j) {
        unsigned long long o = shflx64(best[r][j], mask);
        if (o > best[r][j]) best[r][j] = o;
      }
  if (l16 == 0)
    #pragma unroll
    for (int r = 0; r < 4; ++r)
      #pragma unroll
      for (int j = 0; j < 8; ++j)
        atomicMax(&packed[rg * 64 + r * 16 + j + 8 * hi], best[r][j]);
}

__global__ void sample_k(const unsigned long long* __restrict__ packed,
                         int* __restrict__ tok, float* __restrict__ y, int t) {
  int b = threadIdx.x;                               // 1 block x 256
  unsigned long long p = packed[b];
  int n = (int)(0xFFFFFFFFu - (unsigned)(p & 0xFFFFFFFFull));
  tok[b] = n;
  y[b * SEQ + t] = (float)n;
}

__global__ void eos_k(float* __restrict__ y) {
  y[threadIdx.x * SEQ + MAXLEN] = (float)EOS_IDX;
}

// ---------------- host ----------------
extern "C" void kernel_launch(void* const* d_in, const int* in_sizes, int n_in,
                              void* d_out, int out_size, void* d_ws, size_t ws_size,
                              hipStream_t stream) {
  (void)in_sizes; (void)n_in; (void)out_size; (void)ws_size;
  const float* z      = (const float*)d_in[0];
  const int*   l      = (const int*)  d_in[1];
  const float* emb_w  = (const float*)d_in[2];
  const float* attr_w = (const float*)d_in[3];
  const float* W_ih   = (const float*)d_in[4];
  const float* W_hh   = (const float*)d_in[5];
  const float* b_ih   = (const float*)d_in[6];
  const float* b_hh   = (const float*)d_in[7];
  const float* out_w  = (const float*)d_in[8];
  const float* out_b  = (const float*)d_in[9];

  float* hy = (float*)d_out;                          // (B,31,700)
  float* y  = hy + (size_t)B * SEQ * HID;             // (B,31) tokens as float

  // workspace carve-up
  char* p = (char*)d_ws;
  auto take = [&](size_t bytes) { void* q = p; p += (bytes + 255) & ~size_t(255); return q; };
  _Float16* outw16 = (_Float16*)take((size_t)NVP * KH * 2);      // ~70.8 MB
  _Float16* wih16  = (_Float16*)take((size_t)NGP * KX * 2);
  _Float16* whh16  = (_Float16*)take((size_t)NGP * KH * 2);
  float*    h32    = (float*)   take((size_t)B * HID * 4);
  _Float16* h16    = (_Float16*)take((size_t)B * KH * 2);
  _Float16* x16    = (_Float16*)take((size_t)B * KX * 2);
  float*    gi     = (float*)   take((size_t)B * NGP * 4);
  float*    gh     = (float*)   take((size_t)B * NGP * 4);
  int*      tok    = (int*)     take(B * 4);
  unsigned long long* packed = (unsigned long long*)take(B * 8);

  // per-step keys: jax.random.split(key(42), 30) via threefry on host
  unsigned keys[MAXLEN][2];
  {
    unsigned w[2 * MAXLEN];
    for (int pi = 0; pi < MAXLEN; ++pi) {
      unsigned y0, y1;
      tf2x32(0u, 42u, (unsigned)pi, (unsigned)(pi + MAXLEN), y0, y1);
      w[pi] = y0; w[pi + MAXLEN] = y1;
    }
    for (int j = 0; j < MAXLEN; ++j) { keys[j][0] = w[2*j]; keys[j][1] = w[2*j+1]; }
  }

  // one-time: f16 weight copies (pad) + h0 + tok=SOS
  {
    size_t tot = (size_t)NVP * KH;
    conv_w_k<<<(unsigned)((tot + 255) / 256), 256, 0, stream>>>(out_w, outw16, VOCAB, HID, KH, tot);
    tot = (size_t)NGP * KX;
    conv_w_k<<<(unsigned)((tot + 255) / 256), 256, 0, stream>>>(W_ih, wih16, NG, EMB, KX, tot);
    tot = (size_t)NGP * KH;
    conv_w_k<<<(unsigned)((tot + 255) / 256), 256, 0, stream>>>(W_hh, whh16, NG, HID, KH, tot);
    init_h_k<<<(B * KH + 255) / 256, 256, 0, stream>>>(z, l, attr_w, h32, h16, tok);
  }

  const int gemmBlocks = ((B / 16) * NGT + 3) / 4;    // 4 waves/block, 1 tile/wave
  const dim3 lgrid((NVT + 7) / 8, 4);                 // col-tile groups x row-groups
  const size_t lshmem = (size_t)64 * KH * sizeof(_Float16);

  for (int t = 0; t < SEQ; ++t) {
    int fixed = (t == MAXLEN) ? EOS_IDX : -1;
    gather_x_k<<<(B * KX + 255) / 256, 256, 0, stream>>>(emb_w, tok, x16, fixed);
    gemm_k<<<gemmBlocks, 128, 0, stream>>>(x16, wih16, gi, KX, NGT);
    gemm_k<<<gemmBlocks, 128, 0, stream>>>(h16, whh16, gh, KH, NGT);
    gates_k<<<(B * HID + 255) / 256, 256, 0, stream>>>(gi, gh, b_ih, b_hh, h32, h32,
                                                       h16, hy, t, packed);
    if (t < MAXLEN) {
      logits_sample_k<<<lgrid, 128, lshmem, stream>>>(h16, outw16, out_b, packed,
                                                      keys[t][0], keys[t][1]);
      sample_k<<<1, B, 0, stream>>>(packed, tok, y, t);
    }
  }
  eos_k<<<1, B, 0, stream>>>(y);
}
